// net_77360950936193
// MI455X (gfx1250) — compile-verified, matrix operand order checked by reading, not run on hardware
//
#include <hip/hip_runtime.h>
#include <hip/hip_bf16.h>

// ---------- CDNA5 wave32 WMMA types ----------
typedef __attribute__((ext_vector_type(16))) __bf16 v16bf;
typedef __attribute__((ext_vector_type(8)))  __bf16 v8bf;
typedef __attribute__((ext_vector_type(8)))  float  v8f;
typedef int v4i_vs __attribute__((vector_size(16)));   // matches builtin param type

#if __has_builtin(__builtin_amdgcn_global_load_async_to_lds_b128) && \
    __has_builtin(__builtin_amdgcn_s_wait_asynccnt)
#define USE_ASYNC_LDS 1
#else
#define USE_ASYNC_LDS 0
#endif

__device__ __forceinline__ v8f wmma_bf16(v16bf a, v16bf b, v8f c) {
    return __builtin_amdgcn_wmma_f32_16x16x32_bf16(
        false, a, false, b, (short)0, c, false, false);
}

__device__ __forceinline__ v16bf load_frag(const __bf16* p) {
    v16bf f;
    v8bf lo = *(const v8bf*)p;
    v8bf hi = *(const v8bf*)(p + 8);
    #pragma unroll
    for (int i = 0; i < 8; ++i) { f[i] = lo[i]; f[i + 8] = hi[i]; }
    return f;
}

__device__ __forceinline__ v16bf zero_frag() {
    v16bf f;
    #pragma unroll
    for (int i = 0; i < 16; ++i) f[i] = (__bf16)0.0f;
    return f;
}

// ============================================================
// f32 NCHW -> bf16 NHWC
// ============================================================
__global__ void nchw_to_nhwc_bf16(const float* __restrict__ src,
                                  __bf16* __restrict__ dst,
                                  int N, int C, int H, int W) {
    long total = (long)N * C * H * W;
    long idx = (long)blockIdx.x * blockDim.x + threadIdx.x;
    if (idx >= total) return;
    int c = (int)(idx % C);  long t = idx / C;
    int w = (int)(t % W);    t /= W;
    int h = (int)(t % H);
    int n = (int)(t / H);
    dst[idx] = (__bf16)src[((long)(n * C + c) * H + h) * W + w];
}

// ============================================================
// Pack OIHW f32 weights into WMMA-native bf16 tiles:
//   layout: [kt][nt][lane(32)][16 bf16]
//   lane's 16 values: k = kt*32 + (lane>>4)*16 + kk  (K = (kh*KW+kw)*Cin + ci)
//                     col = nt*16 + (lane&15)      zero-padded past Ktot/Cout
// ============================================================
__global__ void pack_weights(const float* __restrict__ w,
                             __bf16* __restrict__ dst,
                             int Cin, int Cout, int K, int Ktiles, int Ntiles) {
    int Ktot = K * K * Cin;
    long total = (long)Ktiles * Ntiles * 512;
    long idx = (long)blockIdx.x * blockDim.x + threadIdx.x;
    if (idx >= total) return;
    int  kk   = (int)(idx & 15);
    int  lane = (int)((idx >> 4) & 31);
    long tile = idx >> 9;
    int  nt   = (int)(tile % Ntiles);
    int  kt   = (int)(tile / Ntiles);
    int  k    = kt * 32 + (lane >> 4) * 16 + kk;
    int  col  = nt * 16 + (lane & 15);
    float v = 0.0f;
    if (k < Ktot && col < Cout) {
        int ci   = k % Cin;
        int khkw = k / Cin;
        int kh   = khkw / K;
        int kw   = khkw % K;
        v = w[((long)col * Cin + ci) * K * K + kh * K + kw];
    }
    dst[idx] = (__bf16)v;
}

// ============================================================
// Implicit-GEMM conv, bf16 WMMA, f32 accumulate.
//  Block = 256 threads = 8 waves = 4 M-waves x 2 N-pairs
//  Block tile: 128 (M) x 64 (N).  Wave tile: 32x32 (4 WMMA accumulators).
//  Weights double-buffered through LDS with CDNA5 async copies:
//    issue stage(kt+1) -> s_wait_asynccnt(1) -> barrier -> compute(kt)
// ============================================================
__global__ __launch_bounds__(256)
void conv_wmma_bf16(const __bf16* __restrict__ act,   // NHWC bf16
                    const __bf16* __restrict__ wpk,   // packed tiles
                    const float*  __restrict__ bias,
                    __bf16* __restrict__ out,         // NHWC bf16
                    int Cin, int H, int W, int Cout, int K, int pad,
                    int OH, int OW, int M, int NtilesP, int Ktiles) {
    __shared__ __bf16 sB[2][4 * 512];   // 2 x 4 KB ping-pong weight stage

    const int tid   = threadIdx.x;
    const int lane  = tid & 31;
    const int wid   = tid >> 5;
    const int mwave = wid & 3;       // 0..3 along M
    const int npair = wid >> 2;      // 0..1 along N
    const int khalf = lane >> 4;

    // ---- per-lane A rows (two 16-row M tiles per wave) ----
    const int mBase = (blockIdx.x * 4 + mwave) * 32;
    const int r0 = mBase + (lane & 15);
    const int r1 = r0 + 16;
    const bool v0 = (r0 < M), v1 = (r1 < M);
    int n0 = 0, oh0 = 0, ow0 = 0, n1 = 0, oh1 = 0, ow1 = 0;
    if (v0) { int r = r0; ow0 = r % OW; r /= OW; oh0 = r % OH; n0 = r / OH; }
    if (v1) { int r = r1; ow1 = r % OW; r /= OW; oh1 = r % OH; n1 = r / OH; }

    const long wTileBase = (long)blockIdx.y * 4;   // first n-tile of this block
    const int  fragOff   = (npair * 2) * 512 + lane * 16;

    v8f a00 = {0,0,0,0,0,0,0,0}, a01 = a00, a10 = a00, a11 = a00;

    // issue cooperative B-stage for K-step kt into buffer buf (256 x 16B = 4KB)
    auto issueStage = [&](int kt, int buf) {
        const __bf16* g = wpk + ((long)kt * NtilesP + wTileBase) * 512 + tid * 8;
        __bf16* d = &sB[buf][tid * 8];
#if USE_ASYNC_LDS
        __builtin_amdgcn_global_load_async_to_lds_b128(
            (__attribute__((address_space(1))) v4i_vs*)g,
            (__attribute__((address_space(3))) v4i_vs*)d, 0, 0);
#else
        *(v8bf*)d = *(const v8bf*)g;
#endif
    };

    auto mma4 = [&](int buf, v16bf A0, v16bf A1) {
        const __bf16* b0 = &sB[buf][fragOff];
        v16bf B0 = load_frag(b0);                  // ds_load_b128 x2
        v16bf B1 = load_frag(b0 + 512);
        a00 = wmma_bf16(A0, B0, a00);
        a01 = wmma_bf16(A0, B1, a01);
        a10 = wmma_bf16(A1, B0, a10);
        a11 = wmma_bf16(A1, B1, a11);
    };

    // pipelined step: prefetch kt+1, retire kt
    auto pipeSync = [&](int kt) {
        int cur = kt & 1;
        if (kt + 1 < Ktiles) {
            issueStage(kt + 1, cur ^ 1);
#if USE_ASYNC_LDS
            __builtin_amdgcn_s_wait_asynccnt(1);   // stage(kt) landed; kt+1 in flight
#endif
        } else {
#if USE_ASYNC_LDS
            __builtin_amdgcn_s_wait_asynccnt(0);
#endif
        }
        __syncthreads();                           // buf[cur] visible to all waves
        return cur;
    };

    issueStage(0, 0);                              // prologue

    if ((Cin & 31) == 0) {
        // ---- vector path: explicit (kh, kw, ci-block) nest; no inner div ----
        const int CB = Cin >> 5;
        int kt = 0;
        for (int kh = 0; kh < K; ++kh) {
            const int ih0 = oh0 - pad + kh;
            const int ih1 = oh1 - pad + kh;
            for (int kw = 0; kw < K; ++kw) {
                const int iw0 = ow0 - pad + kw;
                const int iw1 = ow1 - pad + kw;
                const bool in0 = v0 && ih0 >= 0 && ih0 < H && iw0 >= 0 && iw0 < W;
                const bool in1 = v1 && ih1 >= 0 && ih1 < H && iw1 >= 0 && iw1 < W;
                const __bf16* p0 = act + ((long)(n0 * H + ih0) * W + iw0) * Cin + khalf * 16;
                const __bf16* p1 = act + ((long)(n1 * H + ih1) * W + iw1) * Cin + khalf * 16;
                for (int cb = 0; cb < CB; ++cb, ++kt) {
                    v16bf A0 = in0 ? load_frag(p0 + cb * 32) : zero_frag();
                    v16bf A1 = in1 ? load_frag(p1 + cb * 32) : zero_frag();
                    int cur = pipeSync(kt);
                    mma4(cur, A0, A1);
                    __syncthreads();               // reads done before buf reuse
                }
            }
        }
    } else {
        // ---- scalar gather path (conv1: Cin=3, single K-tile of 27) ----
        const int Ktot = K * K * Cin;
        for (int kt = 0; kt < Ktiles; ++kt) {
            const int kbase = kt * 32 + khalf * 16;
            v16bf A0 = zero_frag(), A1 = zero_frag();
            #pragma unroll
            for (int kk = 0; kk < 16; ++kk) {
                int k = kbase + kk;
                if (k < Ktot) {
                    int ci   = k % Cin;
                    int khkw = k / Cin;
                    int kh   = khkw / K;
                    int kw   = khkw % K;
                    int ih0 = oh0 - pad + kh, iw0 = ow0 - pad + kw;
                    int ih1 = oh1 - pad + kh, iw1 = ow1 - pad + kw;
                    if (v0 && ih0 >= 0 && ih0 < H && iw0 >= 0 && iw0 < W)
                        A0[kk] = act[((long)(n0 * H + ih0) * W + iw0) * Cin + ci];
                    if (v1 && ih1 >= 0 && ih1 < H && iw1 >= 0 && iw1 < W)
                        A1[kk] = act[((long)(n1 * H + ih1) * W + iw1) * Cin + ci];
                }
            }
            int cur = pipeSync(kt);
            mma4(cur, A0, A1);
            __syncthreads();
        }
    }

    // ---- bias + store  (C layout: lane l, vgpr r -> M = r + 8*(l>=16), N = l%16) ----
    const int  c0 = blockIdx.y * 64 + npair * 32 + (lane & 15);
    const int  c1 = c0 + 16;
    const float bb0 = (c0 < Cout) ? bias[c0] : 0.0f;
    const float bb1 = (c1 < Cout) ? bias[c1] : 0.0f;
    #pragma unroll
    for (int r = 0; r < 8; ++r) {
        int m = mBase + khalf * 8 + r;
        if (m < M) {
            int t = m; int ww = t % OW; t /= OW; int hh = t % OH; int nn = t / OH;
            __bf16* o = out + ((long)(nn * OH + hh) * OW + ww) * Cout;
            if (c0 < Cout) o[c0] = (__bf16)(a00[r] + bb0);
            if (c1 < Cout) o[c1] = (__bf16)(a01[r] + bb1);
        }
        int m2 = m + 16;
        if (m2 < M) {
            int t = m2; int ww = t % OW; t /= OW; int hh = t % OH; int nn = t / OH;
            __bf16* o = out + ((long)(nn * OH + hh) * OW + ww) * Cout;
            if (c0 < Cout) o[c0] = (__bf16)(a10[r] + bb0);
            if (c1 < Cout) o[c1] = (__bf16)(a11[r] + bb1);
        }
    }
}

// ============================================================
// 2x2 maxpool, stride 2, -inf padding (floor mode), bf16 NHWC
// ============================================================
__global__ void maxpool2_bf16(const __bf16* __restrict__ in,
                              __bf16* __restrict__ out,
                              int N, int C, int H, int W,
                              int p, int OH, int OW) {
    long total = (long)N * OH * OW * C;
    long idx = (long)blockIdx.x * blockDim.x + threadIdx.x;
    if (idx >= total) return;
    int c  = (int)(idx % C);  long t = idx / C;
    int ow = (int)(t % OW);   t /= OW;
    int oh = (int)(t % OH);
    int n  = (int)(t / OH);
    float m = -INFINITY;
    #pragma unroll
    for (int dy = 0; dy < 2; ++dy) {
        #pragma unroll
        for (int dx = 0; dx < 2; ++dx) {
            int ih = oh * 2 - p + dy;
            int iw = ow * 2 - p + dx;
            if (ih >= 0 && ih < H && iw >= 0 && iw < W)
                m = fmaxf(m, (float)in[((long)(n * H + ih) * W + iw) * C + c]);
        }
    }
    out[idx] = (__bf16)m;
}

// ============================================================
// Head: transpose [B,6,H,W]->[B,W,H,6] + IoU-masked L1 loss
// ============================================================
__global__ void zero_loss(float* p) { *p = 0.0f; }

__global__ void head_kernel(const __bf16* __restrict__ act, // NHWC [B,S,S,6]
                            float* __restrict__ out,        // [B,S,S,6] + loss
                            int B, int S) {
    int total = B * S * S;
    int idx = blockIdx.x * blockDim.x + threadIdx.x;
    if (idx >= total) return;
    int k = idx % S;  int t = idx / S;   // k indexes H
    int j = t % S;                       // j indexes W
    int b = t / S;

    const __bf16* p = act + ((long)(b * S + k) * S + j) * 6;  // x[b,:,h=k,w=j]
    float v[6];
    #pragma unroll
    for (int c = 0; c < 6; ++c) v[c] = (float)p[c];

    float* o = out + ((long)(b * S + j) * S + k) * 6;         // xt[b,j,k,:]
    #pragma unroll
    for (int c = 0; c < 6; ++c) o[c] = v[c];

    const float gx = 0.567f, gy = 0.469f, gw = 0.206f, gh = 0.53f;
    float fS = (float)S;
    float px = ((float)j + v[2]) / fS;
    float py = ((float)k + v[3]) / fS;
    float pw = __expf(v[4]) / fS;
    float ph = __expf(v[5]) / fS;
    float iw = fminf(px + pw * 0.5f, gx + gw * 0.5f) - fmaxf(px - pw * 0.5f, gx - gw * 0.5f);
    iw = fmaxf(iw, 0.0f);
    float ih = fminf(py + ph * 0.5f, gy + gh * 0.5f) - fmaxf(py - ph * 0.5f, gy - gh * 0.5f);
    ih = fmaxf(ih, 0.0f);
    float inter = iw * ih;
    float iou = inter / (pw * ph + gw * gh - inter);

    float tx = gx * fS - (float)j;
    float ty = gy * fS - (float)k;
    float tw = __logf(gw * fS);
    float th = __logf(gh * fS);
    float l1 = fabsf(tx - v[2]) + fabsf(ty - v[3]) + fabsf(tw - v[4]) + fabsf(th - v[5]);
    float contrib = (iou > 0.5f) ? l1 : 0.0f;
    if (contrib != 0.0f)
        atomicAdd(&out[(long)B * S * S * 6], contrib);
}

// ============================================================
// Host sequencing
// ============================================================
struct CP { int cin, cout, k, pad; };
static const CP g_convs[18] = {
    {3,32,3,0},{32,64,3,0},{64,128,3,1},{128,64,1,0},{64,128,3,1},
    {128,256,3,1},{256,128,1,0},{128,256,3,0},{256,512,3,1},{512,256,1,0},
    {256,512,3,1},{512,256,1,0},{256,512,3,0},{512,256,3,1},{256,128,1,0},
    {128,64,3,1},{64,32,1,0},{32,6,3,1}};

static int pool_pad_after(int i1) {
    switch (i1) { case 1: return 1; case 2: return 1; case 5: return 0;
                  case 8: return 1; case 13: return 1; default: return -1; }
}

extern "C" void kernel_launch(void* const* d_in, const int* in_sizes, int n_in,
                              void* d_out, int out_size, void* d_ws, size_t ws_size,
                              hipStream_t stream) {
    (void)in_sizes; (void)n_in; (void)out_size; (void)ws_size;
    const float* x = (const float*)d_in[0];

    // workspace: two bf16 ping-pong activation buffers + packed weights
    const long ACT_ELEMS = 16L * 32 * 414 * 414;   // largest intermediate (conv1 out)
    __bf16* actA  = (__bf16*)d_ws;
    __bf16* actB  = actA + ACT_ELEMS;
    __bf16* wbase = actB + ACT_ELEMS;

    // input f32 NCHW -> bf16 NHWC
    {
        long total = 16L * 3 * 416 * 416;
        nchw_to_nhwc_bf16<<<dim3((unsigned)((total + 255) / 256)), 256, 0, stream>>>(
            x, actA, 16, 3, 416, 416);
    }

    int H = 416, W = 416;
    __bf16* cur = actA;
    __bf16* nxt = actB;
    __bf16* wp  = wbase;

    for (int i = 0; i < 18; ++i) {
        const CP c = g_convs[i];
        const int K = c.k, pad = c.pad;
        const int OH = H + 2 * pad - K + 1;
        const int OW = W + 2 * pad - K + 1;
        const int Ktot    = K * K * c.cin;
        const int Ktiles  = (Ktot + 31) / 32;
        const int Nblk    = (c.cout + 63) / 64;     // grid.y (64-col blocks)
        const int NtilesP = Nblk * 4;               // packed n-tiles (zero-padded)
        const long wElems = (long)Ktiles * NtilesP * 512;

        pack_weights<<<dim3((unsigned)((wElems + 255) / 256)), 256, 0, stream>>>(
            (const float*)d_in[1 + 2 * i], wp, c.cin, c.cout, K, Ktiles, NtilesP);

        const int M = 16 * OH * OW;
        dim3 grid((unsigned)((M + 127) / 128), (unsigned)Nblk);
        conv_wmma_bf16<<<grid, 256, 0, stream>>>(
            cur, wp, (const float*)d_in[2 + 2 * i], nxt,
            c.cin, H, W, c.cout, K, pad, OH, OW, M, NtilesP, Ktiles);

        wp += wElems;
        { __bf16* t = cur; cur = nxt; nxt = t; }
        H = OH; W = OW;
        int Cc = c.cout;

        int pp = pool_pad_after(i + 1);
        if (pp >= 0) {
            int PH = (H + 2 * pp) / 2;
            int PW = (W + 2 * pp) / 2;
            long total = (long)16 * PH * PW * Cc;
            maxpool2_bf16<<<dim3((unsigned)((total + 255) / 256)), 256, 0, stream>>>(
                cur, nxt, 16, Cc, H, W, pp, PH, PW);
            { __bf16* t = cur; cur = nxt; nxt = t; }
            H = PH; W = PW;
        }
    }

    // head: cur = [16,13,13,6] bf16 NHWC ; out = xt flat + loss scalar
    float* out = (float*)d_out;
    zero_loss<<<1, 1, 0, stream>>>(out + 16L * 13 * 13 * 6);
    head_kernel<<<dim3((unsigned)((16 * 13 * 13 + 255) / 256)), 256, 0, stream>>>(
        cur, out, 16, 13);
}
